// GAT_19181323944512
// MI455X (gfx1250) — matrix operand order repaired and checked by reading, not compiled, not run on hardware
//
#include <hip/hip_runtime.h>

// ---------------- problem constants (match reference) ----------------
#define N_NODES 50000
#define E_EDGES 800000
#define IN_DIM  128
#define HIDF    64
#define NHEADS  2
#define P_PAIRS 100000
#define SLOPE   0.2f
#define BN_EPS  1e-5f

typedef __attribute__((ext_vector_type(16))) _Float16 v16h;
typedef __attribute__((ext_vector_type(8)))  float    v8f;

// ---------------- helpers ----------------
__device__ __forceinline__ unsigned mono_enc(float f) {
    unsigned u = __float_as_uint(f);
    return (u & 0x80000000u) ? ~u : (u | 0x80000000u);
}
__device__ __forceinline__ float mono_dec(unsigned u) {
    return (u & 0x80000000u) ? __uint_as_float(u & 0x7fffffffu) : __uint_as_float(~u);
}

__global__ void fill_f32(float* __restrict__ p, float v, int n) {
    int i = blockIdx.x * blockDim.x + threadIdx.x;
    if (i < n) p[i] = v;
}
__global__ void fill_u32(unsigned* __restrict__ p, unsigned v, int n) {
    int i = blockIdx.x * blockDim.x + threadIdx.x;
    if (i < n) p[i] = v;
}

// ---------------- operand packing into WMMA f16 fragment layout ----------------
// Fragment chunk id = (tile * kTiles + kt) * 32 + lane; each chunk = 16 halves (32B).
// A fragment (16x32, ISA 7.12.2): lane l, half h -> m = l&15 (clamped), k = (h&7) + (l>>4)*8 + (h>>3)*16
__global__ void pack_a_f16(const float* __restrict__ A, _Float16* __restrict__ Ap,
                           int M, int K, int kTiles)
{
    const int idx = blockIdx.x * blockDim.x + threadIdx.x;      // chunk id
    const int total = ((M + 15) >> 4) * kTiles * 32;
    if (idx >= total) return;
    const int lane = idx & 31;
    const int rest = idx >> 5;
    const int kt = rest % kTiles, tm = rest / kTiles;
    const int l16 = lane & 15, lh = lane >> 4;
    int m = tm * 16 + l16;
    if (m >= M) m = M - 1;   // safety clamp (never hit when M % 16 == 0)
    const float* __restrict__ arow = A + (long)m * K + kt * 32 + lh * 8;
    const float4 a0 = *reinterpret_cast<const float4*>(arow);
    const float4 a1 = *reinterpret_cast<const float4*>(arow + 4);
    const float4 a2 = *reinterpret_cast<const float4*>(arow + 16);
    const float4 a3 = *reinterpret_cast<const float4*>(arow + 20);
    v16h v;
    v[0]  = (_Float16)a0.x; v[1]  = (_Float16)a0.y;
    v[2]  = (_Float16)a0.z; v[3]  = (_Float16)a0.w;
    v[4]  = (_Float16)a1.x; v[5]  = (_Float16)a1.y;
    v[6]  = (_Float16)a1.z; v[7]  = (_Float16)a1.w;
    v[8]  = (_Float16)a2.x; v[9]  = (_Float16)a2.y;
    v[10] = (_Float16)a2.z; v[11] = (_Float16)a2.w;
    v[12] = (_Float16)a3.x; v[13] = (_Float16)a3.y;
    v[14] = (_Float16)a3.z; v[15] = (_Float16)a3.w;
    *reinterpret_cast<v16h*>(Ap + (long)idx * 16) = v;
}

// B fragment (32x16): lane l, half h -> n = l&15, k = (h&7) + (l>>4)*8 + (h>>3)*16
__global__ void pack_b_f16(const float* __restrict__ B, _Float16* __restrict__ Bp,
                           int K, int Nc, int kTiles)
{
    const int idx = blockIdx.x * blockDim.x + threadIdx.x;      // chunk id
    const int total = (Nc >> 4) * kTiles * 32;
    if (idx >= total) return;
    const int lane = idx & 31;
    const int rest = idx >> 5;
    const int kt = rest % kTiles, tn = rest / kTiles;
    const int n = tn * 16 + (lane & 15);
    const int lh = lane >> 4;
    const float* __restrict__ bcol = B + (long)(kt * 32 + lh * 8) * Nc + n;
    v16h v;
#pragma unroll
    for (int h = 0; h < 16; ++h) v[h] = (_Float16)bcol[(long)(h + (h >> 3) * 8) * Nc];
    *reinterpret_cast<v16h*>(Bp + (long)idx * 16) = v;
}

// ---------------- WMMA GEMM on packed fragments ----------------
// C[M x Nc] = A[M x K] @ B[K x Nc]; inner loop = 2 v16h loads + 1 WMMA per K-step.
// CONTRACT: M % 16 == 0 and Nc % 16 == 0 (true for all launches here) -> no
// store-side guards, epilogue is 8 straight coalesced b32 stores per lane.
template <int K>
__global__ __launch_bounds__(128) void gemm_wmma_packed(
        const _Float16* __restrict__ Ap, const _Float16* __restrict__ Bp,
        float* __restrict__ C, int M, int Nc,
        const float* __restrict__ bias, int do_relu)
{
    constexpr int KT = K / 32;
    const int tiles_m = M >> 4;
    const int tiles   = tiles_m * (Nc >> 4);
    const int tile    = blockIdx.x * blockDim.y + threadIdx.y;   // uniform per wave
    if (tile >= tiles) return;                                   // whole-wave exit

    const int tm = tile % tiles_m;
    const int tn = tile / tiles_m;
    const int lane = threadIdx.x;      // wave32
    const int l16  = lane & 15;
    const int lh   = lane >> 4;

    const _Float16* __restrict__ ap = Ap + ((long)tm * KT * 32 + lane) * 16;
    const _Float16* __restrict__ bp = Bp + ((long)tn * KT * 32 + lane) * 16;

    v8f acc = {};
#pragma unroll
    for (int kt = 0; kt < KT; ++kt) {
        const v16h a = *reinterpret_cast<const v16h*>(ap + kt * 32 * 16);
        const v16h b = *reinterpret_cast<const v16h*>(bp + kt * 32 * 16);
        acc = __builtin_amdgcn_wmma_f32_16x16x32_f16(
                  false, a, false, b, (short)0, acc, false, false);
    }

    const int bn = tn * 16 + l16;
    const float bv = bias ? bias[bn] : 0.0f;
    float* __restrict__ crow = C + (long)(tm * 16 + lh * 8) * Nc + bn;
#pragma unroll
    for (int r = 0; r < 8; ++r) {
        // D layout: VGPR r -> row m = tm*16 + r + lh*8, col n = l16
        float v = acc[r] + bv;
        if (do_relu) v = fmaxf(v, 0.0f);
        crow[(long)r * Nc] = v;
    }
}

// ---------------- per-node attention dots: el = <feat,al>, er = <feat,ar> ----------------
__global__ void attn_dots(const float* __restrict__ feat,
                          const float* __restrict__ al, const float* __restrict__ ar,
                          float* __restrict__ el, float* __restrict__ er,
                          int n_nodes, int heads, int fh)
{
    int i = blockIdx.x * blockDim.x + threadIdx.x;   // node*head
    if (i >= n_nodes * heads) return;
    const int n = i / heads, h = i % heads;
    const float* f = feat + (long)n * heads * fh + (long)h * fh;
    const float* a = al + h * fh;
    const float* b = ar + h * fh;
    float sl = 0.f, sr = 0.f;
    for (int k = 0; k < fh; ++k) { sl += f[k] * a[k]; sr += f[k] * b[k]; }
    el[i] = sl;
    er[i] = sr;
}

// ---------------- edge pass 1: per-dst segment max of leaky_relu(el[src]+er[dst]) ----------------
__global__ void edge_max(const float* __restrict__ el, const float* __restrict__ er,
                         const int* __restrict__ src, const int* __restrict__ dst,
                         unsigned* __restrict__ mmax, int n_edges, int heads)
{
    int i = blockIdx.x * blockDim.x + threadIdx.x;
    if (i >= n_edges * heads) return;
    const int e = i / heads, h = i % heads;
    float v = el[src[e] * heads + h] + er[dst[e] * heads + h];
    v = v > 0.f ? v : SLOPE * v;
    atomicMax(&mmax[dst[e] * heads + h], mono_enc(v));
}

// ---------------- edge pass 2: ee = exp(e - max), denom[dst] += ee ----------------
__global__ void edge_exp(const float* __restrict__ el, const float* __restrict__ er,
                         const int* __restrict__ src, const int* __restrict__ dst,
                         const unsigned* __restrict__ mmax,
                         float* __restrict__ ee, float* __restrict__ denom,
                         int n_edges, int heads)
{
    int i = blockIdx.x * blockDim.x + threadIdx.x;
    if (i >= n_edges * heads) return;
    const int e = i / heads, h = i % heads;
    const int d = dst[e];
    float v = el[src[e] * heads + h] + er[d * heads + h];
    v = v > 0.f ? v : SLOPE * v;
    const float m  = mono_dec(mmax[d * heads + h]);
    const float ev = __expf(v - m);
    ee[i] = ev;
    atomicAdd(&denom[d * heads + h], ev);
}

// ---------------- edge pass 3: out[dst] += (ee/denom[dst]) * feat[src] ----------------
__global__ void edge_agg(const float* __restrict__ feat,
                         const int* __restrict__ src, const int* __restrict__ dst,
                         const float* __restrict__ ee, const float* __restrict__ denom,
                         float* __restrict__ out, int n_edges, int F, int fh, int heads)
{
    long i = (long)blockIdx.x * blockDim.x + threadIdx.x;
    if (i >= (long)n_edges * F) return;
    const int e = (int)(i / F), f = (int)(i % F);
    const int h = f / fh;
    const int d = dst[e];
    const float alpha = ee[(long)e * heads + h] / fmaxf(denom[d * heads + h], 1e-9f);
    atomicAdd(&out[(long)d * F + f], alpha * feat[(long)src[e] * F + f]);
}

// ---------------- post-agg bias add + BatchNorm column stats ----------------
__global__ void bias_stats(float* __restrict__ hbuf, const float* __restrict__ bias,
                           float* __restrict__ sum, float* __restrict__ sumsq,
                           int n_nodes, int F)
{
    __shared__ float s1[256], s2[256];
    const int f = blockIdx.x;
    const int n = blockIdx.y * blockDim.x + threadIdx.x;
    float v = 0.f, v2 = 0.f;
    if (n < n_nodes) {
        float t = hbuf[(long)n * F + f] + bias[f];
        hbuf[(long)n * F + f] = t;
        v = t; v2 = t * t;
    }
    s1[threadIdx.x] = v; s2[threadIdx.x] = v2;
    __syncthreads();
    for (int st = 128; st > 0; st >>= 1) {
        if (threadIdx.x < st) { s1[threadIdx.x] += s1[threadIdx.x + st];
                                s2[threadIdx.x] += s2[threadIdx.x + st]; }
        __syncthreads();
    }
    if (threadIdx.x == 0) { atomicAdd(&sum[f], s1[0]); atomicAdd(&sumsq[f], s2[0]); }
}

__global__ void bn_finalize(const float* __restrict__ sum, const float* __restrict__ sumsq,
                            float* __restrict__ mean, float* __restrict__ rstd, int F, float invN)
{
    int f = blockIdx.x * blockDim.x + threadIdx.x;
    if (f >= F) return;
    const float mu  = sum[f] * invN;
    const float var = sumsq[f] * invN - mu * mu;
    mean[f] = mu;
    rstd[f] = rsqrtf(var + BN_EPS);
}

__global__ void bn_relu_apply(float* __restrict__ hbuf,
                              const float* __restrict__ mean, const float* __restrict__ rstd,
                              const float* __restrict__ g, const float* __restrict__ b,
                              int n_nodes, int F)
{
    int i = blockIdx.x * blockDim.x + threadIdx.x;
    if (i >= n_nodes * F) return;
    const int f = i % F;
    const float v = g[f] * (hbuf[i] - mean[f]) * rstd[f] + b[f];
    hbuf[i] = fmaxf(v, 0.0f);
}

__global__ void add_bias(float* __restrict__ hbuf, const float* __restrict__ bias,
                         int n_nodes, int F)
{
    int i = blockIdx.x * blockDim.x + threadIdx.x;
    if (i >= n_nodes * F) return;
    hbuf[i] += bias[i % F];
}

// ---------------- predictor: z = h[src]*h[dst] gather-multiply ----------------
__global__ void pair_mul(const float* __restrict__ hbuf,
                         const int* __restrict__ s, const int* __restrict__ d,
                         float* __restrict__ z, int npairs, int F)
{
    int i = blockIdx.x * blockDim.x + threadIdx.x;
    if (i >= npairs * F) return;
    const int p = i / F, f = i % F;
    z[i] = hbuf[(long)s[p] * F + f] * hbuf[(long)d[p] * F + f];
}

// ---------------- predictor final: out[p] = <z[p,:], w> + b ----------------
__global__ void final_dot(const float* __restrict__ z, const float* __restrict__ w,
                          const float* __restrict__ b, float* __restrict__ out, int npairs, int F)
{
    int p = blockIdx.x * blockDim.x + threadIdx.x;
    if (p >= npairs) return;
    const float* zr = z + (long)p * F;
    float s = 0.f;
    for (int f = 0; f < F; ++f) s += zr[f] * w[f];
    out[p] = s + b[0];
}

// ---------------- host-side orchestration ----------------
static inline int cdiv(long a, long b) { return (int)((a + b - 1) / b); }

extern "C" void kernel_launch(void* const* d_in, const int* in_sizes, int n_in,
                              void* d_out, int out_size, void* d_ws, size_t ws_size,
                              hipStream_t stream)
{
    // inputs in setup_inputs() dict order
    const float* x        = (const float*)d_in[0];
    const int*   edge_src = (const int*)  d_in[1];
    const int*   edge_dst = (const int*)  d_in[2];
    const int*   pos_src  = (const int*)  d_in[3];
    const int*   pos_dst  = (const int*)  d_in[4];
    const int*   neg_src  = (const int*)  d_in[5];
    const int*   neg_dst  = (const int*)  d_in[6];
    const float* W0  = (const float*)d_in[7];
    const float* al0 = (const float*)d_in[8];
    const float* ar0 = (const float*)d_in[9];
    const float* b0  = (const float*)d_in[10];
    const float* W1  = (const float*)d_in[11];
    const float* al1 = (const float*)d_in[12];
    const float* ar1 = (const float*)d_in[13];
    const float* b1  = (const float*)d_in[14];
    const float* W2  = (const float*)d_in[15];
    const float* al2 = (const float*)d_in[16];
    const float* ar2 = (const float*)d_in[17];
    const float* b2  = (const float*)d_in[18];
    const float* bn0_g = (const float*)d_in[19];
    const float* bn0_b = (const float*)d_in[20];
    const float* bn1_g = (const float*)d_in[21];
    const float* bn1_b = (const float*)d_in[22];
    const float* pW1 = (const float*)d_in[23];
    const float* pb1 = (const float*)d_in[24];
    const float* pW2 = (const float*)d_in[25];
    const float* pb2 = (const float*)d_in[26];
    const float* pW3 = (const float*)d_in[27];
    const float* pb3 = (const float*)d_in[28];
    float* out = (float*)d_out;

    // workspace layout (float units; all offsets multiples of 8 -> 32B aligned)
    float* ws = (float*)d_ws;
    const long F0 = NHEADS * HIDF;                 // 128
    float*     featA = ws;                          // N x 128
    float*     featB = featA + (long)N_NODES * F0;  // N x 128 (h buffer)
    float*     el    = featB + (long)N_NODES * F0;  // N x 2
    float*     er    = el + (long)N_NODES * NHEADS;
    unsigned*  mmax  = (unsigned*)(er + (long)N_NODES * NHEADS);
    float*     denom = (float*)(mmax + (long)N_NODES * NHEADS);
    float*     ee    = denom + (long)N_NODES * NHEADS;   // E x 2
    float*     bsum  = ee + (long)E_EDGES * NHEADS;      // 128
    float*     bssq  = bsum + F0;
    float*     bmean = bssq + F0;
    float*     brstd = bmean + F0;
    float*     z1    = brstd + F0;                       // P x 64
    float*     z2    = z1 + (long)P_PAIRS * HIDF;        // P x 64
    _Float16*  packA = (_Float16*)(z2 + (long)P_PAIRS * HIDF);  // max(N*128, P*64) halves
    _Float16*  packB = packA + (long)N_NODES * F0;              // K*Nc halves (<=16384)

    const unsigned NEG_INF_ENC = 0x007fffffu;  // mono_enc(-inf)
    const int TB = 256;

    // one GAT layer (all fc layers have K = 128): featA = hin @ W, then
    // softmax-attention aggregate into outbuf
    auto gat_layer = [&](const float* hin, const float* W,
                         const float* al, const float* ar,
                         int heads, int fh, float* outbuf) {
        const int F = heads * fh;
        const int kT = 128 / 32;
        // pack operands into WMMA fragment layout (f32 -> f16 once)
        const int aChunks = cdiv(N_NODES, 16) * kT * 32;
        pack_a_f16<<<cdiv(aChunks, TB), TB, 0, stream>>>(hin, packA, N_NODES, 128, kT);
        const int bChunks = (F / 16) * kT * 32;
        pack_b_f16<<<cdiv(bChunks, TB), TB, 0, stream>>>(W, packB, 128, F, kT);
        // fc (no bias — reference adds bias post-aggregation)
        const int tiles = cdiv(N_NODES, 16) * (F / 16);
        gemm_wmma_packed<128><<<cdiv(tiles, 4), dim3(32, 4), 0, stream>>>(
            packA, packB, featA, N_NODES, F, nullptr, 0);
        attn_dots<<<cdiv((long)N_NODES * heads, TB), TB, 0, stream>>>(
            featA, al, ar, el, er, N_NODES, heads, fh);
        fill_u32<<<cdiv((long)N_NODES * heads, TB), TB, 0, stream>>>(
            mmax, NEG_INF_ENC, N_NODES * heads);
        fill_f32<<<cdiv((long)N_NODES * heads, TB), TB, 0, stream>>>(
            denom, 0.f, N_NODES * heads);
        edge_max<<<cdiv((long)E_EDGES * heads, TB), TB, 0, stream>>>(
            el, er, edge_src, edge_dst, mmax, E_EDGES, heads);
        edge_exp<<<cdiv((long)E_EDGES * heads, TB), TB, 0, stream>>>(
            el, er, edge_src, edge_dst, mmax, ee, denom, E_EDGES, heads);
        fill_f32<<<cdiv((long)N_NODES * F, TB), TB, 0, stream>>>(
            outbuf, 0.f, N_NODES * F);
        edge_agg<<<cdiv((long)E_EDGES * F, TB), TB, 0, stream>>>(
            featA, edge_src, edge_dst, ee, denom, outbuf, E_EDGES, F, fh, heads);
    };

    // BN(train-normalization) + ReLU epilogue, bias fused into stats pass
    auto bn_relu = [&](float* hbuf, const float* bias, const float* g, const float* bb) {
        fill_f32<<<1, 256, 0, stream>>>(bsum, 0.f, 2 * (int)F0);  // bsum+bssq contiguous
        dim3 grid((int)F0, cdiv(N_NODES, TB));
        bias_stats<<<grid, TB, 0, stream>>>(hbuf, bias, bsum, bssq, N_NODES, (int)F0);
        bn_finalize<<<1, (int)F0, 0, stream>>>(bsum, bssq, bmean, brstd, (int)F0, 1.0f / N_NODES);
        bn_relu_apply<<<cdiv((long)N_NODES * F0, TB), TB, 0, stream>>>(
            hbuf, bmean, brstd, g, bb, N_NODES, (int)F0);
    };

    // ---- layer 0: x -> featB ----
    gat_layer(x, W0, al0, ar0, NHEADS, HIDF, featB);
    bn_relu(featB, b0, bn0_g, bn0_b);
    // ---- layer 1: featB -> featB ----
    gat_layer(featB, W1, al1, ar1, NHEADS, HIDF, featB);
    bn_relu(featB, b1, bn1_g, bn1_b);
    // ---- layer 2 (1 head, no BN): featB -> featB (N x 64) ----
    gat_layer(featB, W2, al2, ar2, 1, HIDF, featB);
    add_bias<<<cdiv((long)N_NODES * HIDF, TB), TB, 0, stream>>>(featB, b2, N_NODES, HIDF);

    // ---- link predictor (K = 64), run for pos then neg pairs ----
    auto predict = [&](const int* s, const int* d, float* outp) {
        const int kT = 64 / 32;
        pair_mul<<<cdiv((long)P_PAIRS * HIDF, TB), TB, 0, stream>>>(
            featB, s, d, z1, P_PAIRS, HIDF);
        const int tiles = cdiv(P_PAIRS, 16) * (HIDF / 16);
        const int aChunks = cdiv(P_PAIRS, 16) * kT * 32;
        const int bChunks = (HIDF / 16) * kT * 32;
        // z1 @ pW1 -> z2 (+bias, relu)
        pack_a_f16<<<cdiv(aChunks, TB), TB, 0, stream>>>(z1, packA, P_PAIRS, HIDF, kT);
        pack_b_f16<<<cdiv(bChunks, TB), TB, 0, stream>>>(pW1, packB, HIDF, HIDF, kT);
        gemm_wmma_packed<64><<<cdiv(tiles, 4), dim3(32, 4), 0, stream>>>(
            packA, packB, z2, P_PAIRS, HIDF, pb1, 1);
        // z2 @ pW2 -> z1 (+bias, relu)
        pack_a_f16<<<cdiv(aChunks, TB), TB, 0, stream>>>(z2, packA, P_PAIRS, HIDF, kT);
        pack_b_f16<<<cdiv(bChunks, TB), TB, 0, stream>>>(pW2, packB, HIDF, HIDF, kT);
        gemm_wmma_packed<64><<<cdiv(tiles, 4), dim3(32, 4), 0, stream>>>(
            packA, packB, z1, P_PAIRS, HIDF, pb2, 1);
        final_dot<<<cdiv(P_PAIRS, TB), TB, 0, stream>>>(z1, pW3, pb3, outp, P_PAIRS, HIDF);
    };
    predict(pos_src, pos_dst, out);
    predict(neg_src, neg_dst, out + P_PAIRS);

    (void)in_sizes; (void)n_in; (void)out_size; (void)ws_size;
}
